// Encoder_89163521065800
// MI455X (gfx1250) — matrix operand order repaired and checked by reading, not compiled, not run on hardware
//
#include <hip/hip_runtime.h>

#define VOCAB 32000
#define EMB   256
#define HID   256
#define BSZ   64
#define TT    2048

typedef _Float16 v16h __attribute__((ext_vector_type(16)));
typedef float    v8f  __attribute__((ext_vector_type(8)));
typedef unsigned u32x4 __attribute__((ext_vector_type(4)));
typedef int      i32x4 __attribute__((ext_vector_type(4)));
typedef int      i32x8 __attribute__((ext_vector_type(8)));

#if defined(__HIP_DEVICE_COMPILE__) && __has_builtin(__builtin_amdgcn_tensor_load_to_lds)
#define HAVE_TDM 1
#else
#define HAVE_TDM 0
#endif

// Native V_TANH_F32 (CDNA5 trans op) when the toolchain exposes it.
__device__ __forceinline__ float fast_tanh(float x) {
#if defined(__HIP_DEVICE_COMPILE__) && __has_builtin(__builtin_amdgcn_tanhf)
  return __builtin_amdgcn_tanhf(x);
#elif defined(__HIP_DEVICE_COMPILE__) && __has_builtin(__builtin_amdgcn_tanh_f32)
  return __builtin_amdgcn_tanh_f32(x);
#else
  return tanhf(x);
#endif
}

// ---------------------------------------------------------------------------
// TDM helper: DMA `ndw` contiguous dwords from global into LDS.
// D# per CDNA5 ISA: group0 = {count=1, lds_addr,
// global_addr, type=2}; group1 = {data_size=4B, tensor_dim0=ndw,
// tensor_dim1=1, tile_dim0=ndw, tile_dim1=1, stride0=ndw}. ndw < 65536.
// ---------------------------------------------------------------------------
__device__ __forceinline__ void tdm_load_tile(const void* gsrc, unsigned lds_byte,
                                              unsigned ndw) {
#if HAVE_TDM
  unsigned long long ga = (unsigned long long)gsrc;
  u32x4 g0 = { 1u,                                    // count = 1 (valid D#)
               lds_byte,                              // lds_addr (bytes)
               (unsigned)(ga & 0xFFFFFFFFu),          // global_addr[31:0]
               (unsigned)((ga >> 32) & 0x01FFFFFFu)   // global_addr[56:32]
                 | (2u << 30) };                      // type = 2 ("image")
  i32x8 g1 = { (int)(2u << 16),                        // wg_mask=0, data_size=4B
               (int)((ndw & 0xFFFFu) << 16),           // tensor_dim0[15:0]
               (int)(((ndw >> 16) & 0xFFFFu) | (1u << 16)), // dim0 hi | dim1=1
               (int)((ndw & 0xFFFFu) << 16),           // tile_dim0 = ndw
               1,                                      // tile_dim1=1, tile_dim2=0
               (int)ndw,                               // tensor_dim0_stride lo32
               0, 0 };
  i32x4 z4 = { 0, 0, 0, 0 };
#if __has_include(<hip/amd_detail/amd_gfx1250_TDM.h>)
  i32x8 z8 = { 0, 0, 0, 0, 0, 0, 0, 0 };
  __builtin_amdgcn_tensor_load_to_lds(g0, g1, z4, z4, z8, 0);   // clang-23 form
#else
  __builtin_amdgcn_tensor_load_to_lds(g0, g1, z4, z4, 0);       // ROCm 7.2 form
#endif
#else
  (void)gsrc; (void)lds_byte; (void)ndw;
#endif
}

__device__ __forceinline__ void tdm_wait_all() {
#if HAVE_TDM
  __builtin_amdgcn_s_wait_tensorcnt(0);
#endif
}

// ---------------------------------------------------------------------------
// Kernel 0: pack W_ih and W_hh (f32 row-major [K=256][N=256]) into f16
// WMMA-B operand chunk images: chunk c = (ktile*16 + ntile)*32 + lane holds
// 16 halves = W[ktile*32 + (lane>=16?8:0) + h + (h&8)][ntile*16 + (lane&15)].
// 4096 chunks * 32 B = 128 KiB per matrix. 8192 threads, one chunk each.
// ---------------------------------------------------------------------------
__global__ __launch_bounds__(256, 1) void rnn_pack_w(
    const float* __restrict__ Wih, const float* __restrict__ Whh,
    _Float16* __restrict__ WihP, _Float16* __restrict__ WhhP) {
  const int c  = blockIdx.x * 256 + threadIdx.x;   // 0..8191 (blocks 0-15: Wih)
  const int cc = c & 4095;
  const float* __restrict__ W = (c < 4096) ? Wih : Whh;
  _Float16* __restrict__  P   = (c < 4096) ? WihP : WhhP;
  const int ktile = cc >> 9;
  const int ntile = (cc >> 5) & 15;
  const int l     = cc & 31;
  const int n     = ntile * 16 + (l & 15);
  const int kb    = ktile * 32 + ((l & 16) ? 8 : 0);
  v16h v;
  #pragma unroll
  for (int h = 0; h < 16; ++h) v[h] = (_Float16)W[(kb + h + (h & 8)) * HID + n];
  *(v16h*)(P + cc * 16) = v;
}

// ---------------------------------------------------------------------------
// Kernel 1: X[t][b][:] = f16( embedding[source[b][t]] @ W_ih + b_ih )
// 8-wave blocks; packed W_ih TDM-loaded once into LDS (128 KiB), so each
// B-operand fill is one 32 B ds load. A comes from the embedding gather
// (coalesced float4 loads + cvt). One wave = 16 rows x 64 cols, 32 WMMAs.
// ---------------------------------------------------------------------------
__global__ __launch_bounds__(256, 1) void rnn_xgemm(
    const int* __restrict__ src, const float* __restrict__ emb,
    const _Float16* __restrict__ WihP, const float* __restrict__ bih,
    _Float16* __restrict__ X) {
  extern __shared__ char smem[];
  _Float16* wih_s = (_Float16*)smem;               // 65536 halves

  const int tid  = threadIdx.x;
  const int lane = tid & 31;
  const int wave = tid >> 5;
  const int q    = lane & 15;
  const int hi   = (lane & 16) ? 1 : 0;

#if HAVE_TDM
  if (wave == 0)
    tdm_load_tile(WihP, (unsigned)__builtin_amdgcn_groupstaticsize(), 32768u);
#else
  { const uint4* s4 = (const uint4*)WihP; uint4* d4 = (uint4*)wih_s;
    for (int i = tid; i < 8192; i += 256) d4[i] = s4[i]; }
#endif

  const int job     = blockIdx.x * 8 + wave;       // 32768 jobs total
  const int rowtile = job >> 2;                    // 16 rows of B*T each
  const int ng0     = (job & 3) * 4;               // first global ntile (of 16)
  const int m0      = rowtile * 16 + q;            // row index (both half-waves)
  const float* __restrict__ ap = emb + (long)src[m0] * EMB;

#if HAVE_TDM
  if (wave == 0) tdm_wait_all();
#endif
  __syncthreads();                                  // W_ih staged in LDS

  const v16h* wv = (const v16h*)wih_s;
  const v8f z8 = { 0, 0, 0, 0, 0, 0, 0, 0 };
  v8f acc[4] = { z8, z8, z8, z8 };

  for (int kt = 0; kt < 8; ++kt) {
    // A operand (16x32 f16): halves 0-7 -> K=kb..kb+7, 8-15 -> K=kb+16..kb+23
    const int kb = kt * 32 + hi * 8;
    const float4 f0 = *(const float4*)(ap + kb);
    const float4 f1 = *(const float4*)(ap + kb + 4);
    const float4 f2 = *(const float4*)(ap + kb + 16);
    const float4 f3 = *(const float4*)(ap + kb + 20);
    v16h a;
    a[0]=(_Float16)f0.x;  a[1]=(_Float16)f0.y;  a[2]=(_Float16)f0.z;  a[3]=(_Float16)f0.w;
    a[4]=(_Float16)f1.x;  a[5]=(_Float16)f1.y;  a[6]=(_Float16)f1.z;  a[7]=(_Float16)f1.w;
    a[8]=(_Float16)f2.x;  a[9]=(_Float16)f2.y;  a[10]=(_Float16)f2.z; a[11]=(_Float16)f2.w;
    a[12]=(_Float16)f3.x; a[13]=(_Float16)f3.y; a[14]=(_Float16)f3.z; a[15]=(_Float16)f3.w;

    #pragma unroll
    for (int c = 0; c < 4; ++c) {
      const v16h b = wv[(kt * 16 + ng0 + c) * 32 + lane];
      acc[c] = __builtin_amdgcn_wmma_f32_16x16x32_f16(
          false, a, false, b, (short)0, acc[c], false, false);
    }
  }

  // C layout: VGPR r -> M = r (+8 upper half-wave), N = lane&15.
  // Store time-major: X[(t*BSZ + b)*HID + n] as f16.
  #pragma unroll
  for (int c = 0; c < 4; ++c) {
    const int n = (ng0 + c) * 16 + q;
    const float bi = bih[n];
    #pragma unroll
    for (int r = 0; r < 8; ++r) {
      const int mm   = rowtile * 16 + r + hi * 8;   // m = b*T + t
      const int bidx = mm >> 11;
      const int tidx = mm & (TT - 1);
      X[((long)tidx * BSZ + bidx) * HID + n] = (_Float16)(acc[c][r] + bi);
    }
  }
}

// ---------------------------------------------------------------------------
// Kernel 2: persistent single-WGP recurrence.
//   h = tanh( X[t] + h @ W_hh + b_hh ),  t = 0..2047
// LDS (dynamic, 229376 B):
//   [0,      131072) packed W_hh (TDM-loaded, WMMA-B chunk layout)
//   [131072, 163840) h f16, swizzled into per-lane WMMA-A chunks
//   [163840, 229376) X double buffer (2 x 32768 B), streamed by the TDM
// 8 waves: wave w owns M-tile (w&3), N-tiles (w>>2)*8 .. +7 -> 64 WMMA / step.
// ---------------------------------------------------------------------------
__global__ __launch_bounds__(256, 1) void rnn_recurrence(
    const _Float16* __restrict__ X, const _Float16* __restrict__ WhhP,
    const float* __restrict__ bhh, float* __restrict__ out) {
  extern __shared__ char smem[];
  _Float16* whh_s = (_Float16*)smem;              // 65536 halves
  _Float16* h_s   = (_Float16*)(smem + 131072);   // 16384 halves
  _Float16* xb    = (_Float16*)(smem + 163840);   // 2 x 16384 halves

  const int tid  = threadIdx.x;
  const int lane = tid & 31;
  const int wave = tid >> 5;
  const int q    = lane & 15;
  const int hi   = (lane & 16) ? 1 : 0;

#if HAVE_TDM
  const unsigned lds0 = (unsigned)__builtin_amdgcn_groupstaticsize();
  if (wave == 0) {
    tdm_load_tile(WhhP, lds0, 32768u);             // packed W_hh -> whh_s
    tdm_load_tile(X, lds0 + 163840u, 8192u);       // x[0] -> buffer 0
  }
#else
  { const uint4* s4 = (const uint4*)WhhP; uint4* d4 = (uint4*)whh_s;
    for (int i = tid; i < 8192; i += 256) d4[i] = s4[i]; }
#endif

  // h0 = 0 (16384 halves)
  for (int i = tid; i < 8192; i += 256) ((unsigned*)h_s)[i] = 0u;

  const int mtile = wave & 3;
  const int nbase = (wave >> 2) * 8;
  float bias[8];
  #pragma unroll
  for (int j = 0; j < 8; ++j) bias[j] = bhh[(nbase + j) * 16 + q];

  const v16h* whhv = (const v16h*)whh_s;
  const v16h* hv   = (const v16h*)h_s;

  for (int t = 0; t < TT; ++t) {
#if HAVE_TDM
    if (wave == 0) tdm_wait_all();
    __syncthreads();   // x[t] + W_hh landed; h stores from t-1 visible
    if (wave == 0 && t + 1 < TT)
      tdm_load_tile(X + (size_t)(t + 1) * (BSZ * HID),
                    lds0 + 163840u + (unsigned)(((t + 1) & 1) * 32768), 8192u);
    const _Float16* xt = xb + (t & 1) * (BSZ * HID);
#else
    __syncthreads();
    {  // fallback: cooperative 128-bit copy of X[t] into LDS
      const uint4* s4 = (const uint4*)(X + (size_t)t * (BSZ * HID));
      uint4* d4 = (uint4*)xb;
      for (int i = tid; i < 2048; i += 256) d4[i] = s4[i];
    }
    __syncthreads();
    const _Float16* xt = xb;
#endif

    // Accumulators seeded with x-projection + b_hh (C layout).
    v8f acc[8];
    #pragma unroll
    for (int j = 0; j < 8; ++j) {
      const int n = (nbase + j) * 16 + q;
      #pragma unroll
      for (int r = 0; r < 8; ++r) {
        const int mm = mtile * 16 + r + hi * 8;
        acc[j][r] = (float)xt[mm * HID + n] + bias[j];
      }
    }

    // h @ W_hh : 8 k-steps x 8 n-tiles of v_wmma_f32_16x16x32_f16.
    #pragma unroll
    for (int kt = 0; kt < 8; ++kt) {
      const v16h a = hv[(mtile * 8 + kt) * 32 + lane];
      #pragma unroll
      for (int j = 0; j < 8; ++j) {
        const v16h b = whhv[(kt * 16 + nbase + j) * 32 + lane];
        acc[j] = __builtin_amdgcn_wmma_f32_16x16x32_f16(
            false, a, false, b, (short)0, acc[j], false, false);
      }
    }

    __syncthreads();   // everyone done reading old h before it is overwritten

    // tanh (native V_TANH_F32), scatter new h into WMMA-A chunk layout (f16).
    #pragma unroll
    for (int j = 0; j < 8; ++j) {
      const int ntile = nbase + j;
      #pragma unroll
      for (int r = 0; r < 8; ++r) {
        const float v = fast_tanh(acc[j][r]);
        const int lr = r + hi * 8;                 // local row 0..15
        const int lp = lr + ((q & 8) ? 16 : 0);    // target lane in A chunk
        const int hp = (q & 7) + (ntile & 1) * 8;  // target half in A chunk
        h_s[((mtile * 8 + (ntile >> 1)) * 32 + lp) * 16 + hp] = (_Float16)v;
        if (t == TT - 1)
          out[(mtile * 16 + lr) * HID + ntile * 16 + q] = v;  // final h_T f32
      }
    }
  }
}

// ---------------------------------------------------------------------------
extern "C" void kernel_launch(void* const* d_in, const int* in_sizes, int n_in,
                              void* d_out, int out_size, void* d_ws, size_t ws_size,
                              hipStream_t stream) {
  (void)in_sizes; (void)n_in; (void)out_size; (void)ws_size;
  const int*   src = (const int*)  d_in[0];
  const float* emb = (const float*)d_in[1];
  const float* Wih = (const float*)d_in[2];
  const float* Whh = (const float*)d_in[3];
  const float* bih = (const float*)d_in[4];
  const float* bhh = (const float*)d_in[5];
  float* out = (float*)d_out;

  // Workspace layout: X [T][B][HID] f16 (64 MiB), then packed Wih / Whh
  // (128 KiB each, WMMA-B chunk images).
  _Float16* X    = (_Float16*)d_ws;
  _Float16* WihP = X + (size_t)TT * BSZ * HID;
  _Float16* WhhP = WihP + 65536;

  rnn_pack_w<<<dim3(32), dim3(256), 0, stream>>>(Wih, Whh, WihP, WhhP);
  rnn_xgemm<<<dim3(4096), dim3(256), 131072, stream>>>(src, emb, WihP, bih, X);
  rnn_recurrence<<<dim3(1), dim3(256), 229376, stream>>>(X, WhhP, bhh, out);
}
